// ScratchNNConv_89275190215166
// MI455X (gfx1250) — compile-verified, matrix operand order checked by reading, not run on hardware
//
#include <hip/hip_runtime.h>

typedef __attribute__((ext_vector_type(16))) _Float16 v16h;
typedef __attribute__((ext_vector_type(8)))  float    v8f;
typedef __attribute__((ext_vector_type(2)))  float    v2f;

#define NN      1024
#define IN_CH   128
#define EDGE_CH 32
#define OUT_CH  128

__device__ __forceinline__ float relu1(float x) {
  // single v_med3_f32: clamp(x, 0, +big) == relu for finite x
  return __builtin_amdgcn_fmed3f(x, 0.0f, 3.0e38f);
}

// ---------------------------------------------------------------------------
// Kernel A: T = node_mat @ node_weight ; R = node_mat @ root   (tiny, VALU)
// ---------------------------------------------------------------------------
__global__ void __launch_bounds__(256) prep_TR_kernel(
    const float* __restrict__ node_mat,
    const float* __restrict__ node_weight,
    const float* __restrict__ root,
    float* __restrict__ T, float* __restrict__ R) {
  int idx = blockIdx.x * blockDim.x + threadIdx.x;   // 0 .. 1024*128-1
  int n = idx >> 7;
  int o = idx & 127;
  const float* nm = node_mat + n * IN_CH;
  float t = 0.f, r = 0.f;
#pragma unroll 8
  for (int i = 0; i < IN_CH; ++i) {
    float x = nm[i];
    t = fmaf(x, node_weight[i * OUT_CH + o], t);
    r = fmaf(x, root[i * OUT_CH + o], r);
  }
  T[idx] = t;
  R[idx] = r;
}

// ---------------------------------------------------------------------------
// Kernel B: g[n,b] = sum_m relu( sum_e edge_adj[e,n,m] * L1[e,b] ) * adj[n,m]
// One wave per node n. Per 32-m double tile:
//   - 2x8 compute WMMAs:  [16m x 32e] x [32e x 128b]
//   - relu(D) repacked IN-LANE into a 32x16 f16 B operand (C-layout row
//     r+8*hi at col lane%16 == B-layout K=base(v,h)+8*hi at col lane%16)
//   - 8 reduction WMMAs: A = broadcast rows of adj weights w[0..31],
//     C-chained f32 accumulator => g without any VALU reduction or shuffles.
// edge_adj (128 MB) streamed exactly once -> HBM-bound (~5.7us floor).
// ---------------------------------------------------------------------------
__global__ void __launch_bounds__(256) edge_g_kernel(
    const float* __restrict__ edge_adj,
    const float* __restrict__ adj,
    const float* __restrict__ L1,     // [32, 127]
    float* __restrict__ g) {          // [1024, 128] (col 127 == 0)
  const int lane = threadIdx.x & 31;
  const int n    = blockIdx.x * 8 + (threadIdx.x >> 5);   // 128 blocks * 8 waves
  const int l    = lane & 15;
  const int hi   = lane >> 4;          // half-wave: K offset +8

  // Resident B operand: L1 padded to [32 x 128] f16, 8 column blocks of 16.
  v16h B[8];
#pragma unroll
  for (int cb = 0; cb < 8; ++cb) {
    int col = cb * 16 + l;
#pragma unroll
    for (int v = 0; v < 8; ++v) {
#pragma unroll
      for (int h = 0; h < 2; ++h) {
        int base = (v < 4) ? (v * 2 + h) : ((v - 4) * 2 + h + 16);
        int k    = base + hi * 8;
        float val = (col < 127) ? L1[k * 127 + col] : 0.0f;
        B[cb][v * 2 + h] = (_Float16)val;
      }
    }
  }

  v8f gacc[8];
#pragma unroll
  for (int cb = 0; cb < 8; ++cb) gacc[cb] = (v8f){};

  const float* adjrow = adj + (size_t)n * NN;
  const size_t plane  = (size_t)NN * NN;
  const float* eabase = edge_adj + (size_t)n * NN;

  for (int mt2 = 0; mt2 < 32; ++mt2) {
    const int mb = mt2 * 32;

    // A operands: edge_adj[:, n, mb..mb+15] and [mb+16..mb+31] as f16 16x32.
    v16h A1, A2;
#pragma unroll
    for (int v = 0; v < 8; ++v) {
#pragma unroll
      for (int h = 0; h < 2; ++h) {
        int base = (v < 4) ? (v * 2 + h) : ((v - 4) * 2 + h + 16);
        int k    = base + hi * 8;
        const float* p = eabase + (size_t)k * plane + mb + l;
        A1[v * 2 + h] = (_Float16)p[0];
        A2[v * 2 + h] = (_Float16)p[16];
      }
    }

    // Reduction A operand: every row = the 32 adj weights of this double tile.
    v16h Ar;
#pragma unroll
    for (int v = 0; v < 8; ++v) {
#pragma unroll
      for (int h = 0; h < 2; ++h) {
        int base = (v < 4) ? (v * 2 + h) : ((v - 4) * 2 + h + 16);
        int k    = base + hi * 8;
        Ar[v * 2 + h] = (_Float16)adjrow[mb + k];
      }
    }

#pragma unroll
    for (int cb = 0; cb < 8; ++cb) {
      v8f z = {};
      v8f d1 = __builtin_amdgcn_wmma_f32_16x16x32_f16(false, A1, false, B[cb],
                                                      (short)0, z, false, false);
      v8f d2 = __builtin_amdgcn_wmma_f32_16x16x32_f16(false, A2, false, B[cb],
                                                      (short)0, z, false, false);
      // relu + in-lane repack into 32x16 f16 B operand (rows = m within tile)
      v16h Br;
#pragma unroll
      for (int i = 0; i < 8; ++i) {
        Br[i]     = (_Float16)relu1(d1[i]);
        Br[8 + i] = (_Float16)relu1(d2[i]);
      }
      // gacc[cb] += w_row_vector * relu(h1_tile)   (all 16 rows identical)
      gacc[cb] = __builtin_amdgcn_wmma_f32_16x16x32_f16(false, Ar, false, Br,
                                                        (short)0, gacc[cb],
                                                        false, false);
    }
  }

  if (lane < 16) {
#pragma unroll
    for (int cb = 0; cb < 8; ++cb)
      g[n * 128 + cb * 16 + lane] = gacc[cb][0];   // any row; all identical
  }
}

// ---------------------------------------------------------------------------
// Kernel C: out = adj @ T + g @ L2 + R + bias, via v_wmma_f32_16x16x4_f32.
// One wave per 16x16 output tile (64 x 8 tiles).
// ---------------------------------------------------------------------------
__global__ void __launch_bounds__(256) final_kernel(
    const float* __restrict__ adj,
    const float* __restrict__ T,
    const float* __restrict__ g,
    const float* __restrict__ L2,     // [127, 128]
    const float* __restrict__ R,
    const float* __restrict__ bias,
    float* __restrict__ out) {
  const int lane = threadIdx.x & 31;
  const int wv   = blockIdx.x * 8 + (threadIdx.x >> 5);   // 0..511
  const int nb   = (wv >> 3) * 16;
  const int ob   = (wv & 7) * 16;
  const int l    = lane & 15;
  const int hi   = lane >> 4;
  const int koff = hi * 2;

  v8f acc = {};

  // adj @ T : K = 1024
  for (int k = 0; k < NN; k += 4) {
    const int kk = k + koff;
    v2f a, b;
    a.x = adj[(size_t)(nb + l) * NN + kk];
    a.y = adj[(size_t)(nb + l) * NN + kk + 1];
    b.x = T[kk * OUT_CH + ob + l];
    b.y = T[(kk + 1) * OUT_CH + ob + l];
    acc = __builtin_amdgcn_wmma_f32_16x16x4_f32(false, a, false, b,
                                                (short)0, acc, false, false);
  }

  // g @ L2 : K = 128 (L2 row 127 zero-padded; g col 127 is already 0)
  for (int k = 0; k < 128; k += 4) {
    const int kk = k + koff;
    v2f a, b;
    a.x = g[(nb + l) * 128 + kk];
    a.y = g[(nb + l) * 128 + kk + 1];
    b.x = (kk < 127)     ? L2[kk * OUT_CH + ob + l]       : 0.0f;
    b.y = (kk + 1 < 127) ? L2[(kk + 1) * OUT_CH + ob + l] : 0.0f;
    acc = __builtin_amdgcn_wmma_f32_16x16x4_f32(false, a, false, b,
                                                (short)0, acc, false, false);
  }

  // C/D layout: VGPR r -> row r (lanes 0-15) / row r+8 (lanes 16-31).
#pragma unroll
  for (int r = 0; r < 8; ++r) {
    int row = nb + r + hi * 8;
    int col = ob + l;
    out[row * OUT_CH + col] = acc[r] + R[row * OUT_CH + col] + bias[col];
  }
}

// ---------------------------------------------------------------------------
extern "C" void kernel_launch(void* const* d_in, const int* in_sizes, int n_in,
                              void* d_out, int out_size, void* d_ws, size_t ws_size,
                              hipStream_t stream) {
  const float* node_mat    = (const float*)d_in[0];
  const float* adj         = (const float*)d_in[1];
  const float* edge_adj    = (const float*)d_in[2];
  const float* node_weight = (const float*)d_in[3];
  const float* L1          = (const float*)d_in[4];
  const float* L2          = (const float*)d_in[5];
  const float* root        = (const float*)d_in[6];
  const float* bias        = (const float*)d_in[7];
  float* out = (float*)d_out;

  float* T = (float*)d_ws;            // [1024,128]
  float* R = T + NN * OUT_CH;         // [1024,128]
  float* g = R + NN * OUT_CH;         // [1024,128]

  prep_TR_kernel<<<512, 256, 0, stream>>>(node_mat, node_weight, root, T, R);
  edge_g_kernel<<<128, 256, 0, stream>>>(edge_adj, adj, L1, g);
  final_kernel<<<64, 256, 0, stream>>>(adj, T, g, L2, R, bias, out);
}
